// KWTA_45414984187969
// MI455X (gfx1250) — compile-verified, matrix operand order checked by reading, not run on hardware
//
#include <hip/hip_runtime.h>
#include <stdint.h>

// KWTA with duty-cycle homeostasis, MI455X (gfx1250, wave32).
// Pipeline:
//   A) kwta_select      : radix-select K-th largest per row -> packed mask bits (8 MB)
//   B) kwta_count_wmma  : counts[d] = sum_rows mask  via V_WMMA_I32_16X16X64_IU8 (A = ones)
//   C) kwta_boost       : boost[d] = exp(-(duty*0.99 + 0.01*counts/B - K/D))
//   D) kwta_apply       : out = x * boost * mask   (float4, non-temporal stores)

#define B_ROWS 32768
#define D_COLS 2048
#define K_SEL  40
#define ALPHA  0.01f
#define GAMMA  1.0f
#define TARGETF (40.0f / 2048.0f)

typedef __attribute__((ext_vector_type(4))) float f4;
typedef __attribute__((ext_vector_type(8))) int   v8i;

__device__ __forceinline__ unsigned sortable(float f) {
    // monotonic float -> uint map (order-preserving for all finite values)
    unsigned u = __float_as_uint(f);
    return u ^ (unsigned)(((int)u >> 31) | (int)0x80000000u);
}

// ---------------- Kernel A: per-row K-th-largest via 4-level radix select ---
// One wave32 per row; 64 values per lane held in registers; per-wave 256-bin
// LDS histogram per 8-bit digit level (4 levels = exact 32-bit threshold).
__global__ __launch_bounds__(256) void kwta_select(const float* __restrict__ x,
                                                   unsigned* __restrict__ maskbits) {
    __shared__ unsigned hist[8][256];
    const int lane = threadIdx.x & 31;
    const int wave = threadIdx.x >> 5;
    const int row  = blockIdx.x * 8 + wave;
    unsigned* h = hist[wave];

    unsigned u[64];
    const float* xr = x + (size_t)row * D_COLS;
#pragma unroll
    for (int i = 0; i < 64; ++i) u[i] = sortable(xr[i * 32 + lane]);

    uint64_t act = ~0ull;      // per-lane active-value bitmask
    unsigned Kp  = K_SEL;      // rank remaining within active set
    unsigned T   = 0;          // accumulated threshold bytes

    for (int level = 0; level < 4; ++level) {
        const int shift = 24 - 8 * level;
        for (int j = lane; j < 256; j += 32) h[j] = 0;
        __syncthreads();
#pragma unroll
        for (int i = 0; i < 64; ++i) {
            if ((act >> i) & 1ull)
                atomicAdd(&h[(u[i] >> shift) & 0xFF], 1u);
        }
        __syncthreads();
        // beta = max bin b with C(b) = sum_{b'>=b} h[b'] >= Kp  (C decreasing in b)
        unsigned v[8]; unsigned laneSum = 0;
#pragma unroll
        for (int t = 0; t < 8; ++t) { v[t] = h[lane * 8 + t]; laneSum += v[t]; }
        unsigned acc = laneSum;                       // inclusive suffix over lanes
        for (int off = 1; off < 32; off <<= 1) {
            unsigned o = __shfl_down(acc, off, 32);
            if (lane + off < 32) acc += o;
        }
        unsigned run = acc - laneSum;                 // strictly-higher lanes
        int bestT = -1; unsigned bestC = 0, hBest = 0;
#pragma unroll
        for (int t = 7; t >= 0; --t) {                // descend within lane's 8 bins
            run += v[t];                              // run = C(lane*8 + t)
            if (bestT < 0 && run >= Kp) { bestT = t; bestC = run; hBest = v[t]; }
        }
        unsigned bal = (unsigned)__ballot(bestT >= 0);
        int srcLane  = 31 - __clz((int)bal);          // highest qualifying lane
        int      bT  = __shfl(bestT, srcLane, 32);
        unsigned bC  = __shfl(bestC, srcLane, 32);
        unsigned hB  = __shfl(hBest, srcLane, 32);
        unsigned beta = (unsigned)(srcLane * 8 + bT);
        Kp -= (bC - hB);                              // subtract count strictly above beta
        T = (T << 8) | beta;
        uint64_t nact = 0;
#pragma unroll
        for (int i = 0; i < 64; ++i) {
            if (((act >> i) & 1ull) && (((u[i] >> shift) & 0xFF) == beta))
                nact |= (1ull << i);
        }
        act = nact;
        __syncthreads();
    }

    // Emit packed mask: word i holds cols [32i, 32i+31], bit l = col 32i+l.
    unsigned wa = 0, wb = 0;
#pragma unroll
    for (int i = 0; i < 64; ++i) {
        unsigned bal = (unsigned)__ballot(u[i] >= T);
        if (i == lane)      wa = bal;
        if (i == lane + 32) wb = bal;
    }
    unsigned* mb = maskbits + (size_t)row * 64;
    mb[lane]      = wa;
    mb[lane + 32] = wb;
}

// ---------------- Kernel B: column counts via WMMA IU8 ----------------------
// Block tile: 64 rows x 256 cols of mask bits. Expand to the documented IU8
// B-operand layout (64x16: lanes 0-15 N=lane K=0..15/32..47, lanes 16-31
// N=lane-16 K=16..31/48..63) in LDS, then D = ones(16x64) x mask(64x16):
// every row of D = per-column sum over 64 rows. 16 WMMAs per block.
__global__ __launch_bounds__(256) void kwta_count_wmma(const unsigned* __restrict__ maskbits,
                                                       unsigned* __restrict__ counts) {
    __shared__ unsigned s_bits[512];      // [row 0..63][word 0..7]
    __shared__ int      s_b[16 * 32 * 8]; // [group g][lane][reg i]
    const int t = threadIdx.x;
    const size_t rowBase = (size_t)blockIdx.x * 64;
    const int colTile = blockIdx.y;       // 0..7  (256 cols each)

    // phase 1: stage this tile's mask bits (2 KB)
    for (int idx = t; idx < 512; idx += 256) {
        int r = idx >> 3, w = idx & 7;
        s_bits[idx] = maskbits[(rowBase + r) * 64 + colTile * 8 + w];
    }
    __syncthreads();

    // phase 2: bits -> IU8 B-operand ints. thread -> (lane l, reg i); 4 rows each.
    {
        const int l = t >> 3, i = t & 7;
        const int n     = l & 15;
        const int khalf = (l >> 4) * 16;
        const int rbase = ((i & 3) * 4) + khalf + ((i & 4) ? 32 : 0);
#pragma unroll
        for (int wd = 0; wd < 8; ++wd) {
            unsigned w0 = s_bits[(rbase + 0) * 8 + wd];
            unsigned w1 = s_bits[(rbase + 1) * 8 + wd];
            unsigned w2 = s_bits[(rbase + 2) * 8 + wd];
            unsigned w3 = s_bits[(rbase + 3) * 8 + wd];
#pragma unroll
            for (int gi = 0; gi < 2; ++gi) {
                int g  = wd * 2 + gi;
                int sh = (g * 16 + n) & 31;   // (col>>5)==wd by construction
                int val = (int)(((w0 >> sh) & 1u)      ) |
                          (int)(((w1 >> sh) & 1u) <<  8) |
                          (int)(((w2 >> sh) & 1u) << 16) |
                          (int)(((w3 >> sh) & 1u) << 24);
                s_b[(g << 8) | (l << 3) | i] = val;
            }
        }
    }
    __syncthreads();

    // phase 3: WMMA (EXEC all-ones here, predication only after the matrix op)
    const int lane = t & 31;
    const int wave = t >> 5;
    v8i aOnes;
#pragma unroll
    for (int i = 0; i < 8; ++i) aOnes[i] = 0x01010101;
#pragma unroll
    for (int gi = 0; gi < 2; ++gi) {
        const int g = wave * 2 + gi;
        v8i bvec;
#pragma unroll
        for (int i = 0; i < 8; ++i) bvec[i] = s_b[(g << 8) | (lane << 3) | i];
        v8i c = {};
        v8i d = __builtin_amdgcn_wmma_i32_16x16x64_iu8(false, aOnes, false, bvec,
                                                       c, false, false);
        if (lane < 16)  // D row M=0 lives in c/d reg 0, lanes 0..15 (N = lane)
            atomicAdd(&counts[colTile * 256 + g * 16 + lane], (unsigned)d[0]);
    }
}

// ---------------- Kernel C: duty EMA + boost table --------------------------
__global__ void kwta_boost(const float* __restrict__ duty,
                           const unsigned* __restrict__ counts,
                           float* __restrict__ boost) {
    int d = blockIdx.x * blockDim.x + threadIdx.x;
    if (d < D_COLS) {
        float nd = duty[d] * (1.0f - ALPHA) +
                   ALPHA * ((float)counts[d] * (1.0f / (float)B_ROWS));
        boost[d] = __expf(-GAMMA * (nd - TARGETF));
    }
}

// ---------------- Kernel D: out = x * boost * mask --------------------------
__global__ __launch_bounds__(256) void kwta_apply(const float* __restrict__ x,
                                                  const unsigned* __restrict__ maskbits,
                                                  const float* __restrict__ boost,
                                                  float* __restrict__ out) {
    const size_t i4 = (size_t)blockIdx.x * 256 + threadIdx.x; // index in float4s
    const int row = (int)(i4 >> 9);       // 512 float4 per row
    const int c4  = (int)(i4 & 511);
    const int col = c4 * 4;
    f4 xv = ((const f4*)x)[i4];
    f4 bo = ((const f4*)boost)[c4];       // 8 KB table, L2-resident
    unsigned m = maskbits[(size_t)row * 64 + (col >> 5)] >> (col & 31);
    f4 o;
    o.x = (m & 1u) ? xv.x * bo.x : 0.0f;
    o.y = (m & 2u) ? xv.y * bo.y : 0.0f;
    o.z = (m & 4u) ? xv.z * bo.z : 0.0f;
    o.w = (m & 8u) ? xv.w * bo.w : 0.0f;
    __builtin_nontemporal_store(o, (f4*)out + i4);  // streaming; keep L2 for x/mask
}

extern "C" void kernel_launch(void* const* d_in, const int* in_sizes, int n_in,
                              void* d_out, int out_size, void* d_ws, size_t ws_size,
                              hipStream_t stream) {
    const float* x    = (const float*)d_in[0];
    const float* duty = (const float*)d_in[1];
    float*       out  = (float*)d_out;

    // workspace: [maskbits: B*64 u32 = 8 MB][counts: D u32][boost: D f32]
    unsigned* maskbits = (unsigned*)d_ws;
    unsigned* counts   = maskbits + (size_t)B_ROWS * 64;
    float*    boost    = (float*)(counts + D_COLS);

    hipMemsetAsync(counts, 0, D_COLS * sizeof(unsigned), stream);
    kwta_select<<<B_ROWS / 8, 256, 0, stream>>>(x, maskbits);
    kwta_count_wmma<<<dim3(B_ROWS / 64, D_COLS / 256), 256, 0, stream>>>(maskbits, counts);
    kwta_boost<<<D_COLS / 256, 256, 0, stream>>>(duty, counts, boost);
    kwta_apply<<<(size_t)B_ROWS * (D_COLS / 4) / 256, 256, 0, stream>>>(x, maskbits, boost, out);
}
